// RG_LRU_41068477285010
// MI455X (gfx1250) — matrix operand order repaired and checked by reading, compiled, not run on hardware
//
#include <hip/hip_runtime.h>
#include <hip/hip_bf16.h>
#include <math.h>

// ---------------------------------------------------------------------------
// Problem constants (from reference): B=8, T=4096, D=1024
// ---------------------------------------------------------------------------
constexpr int B_  = 8;
constexpr int T_  = 4096;
constexpr int D_  = 1024;
constexpr int M_  = B_ * T_;        // 32768 rows for all GEMMs
constexpr int NC_ = 8;              // scan chunks per sequence
constexpr int CL_ = T_ / NC_;       // 512 steps per chunk

typedef __bf16 bf16;
typedef bf16  v16bf __attribute__((ext_vector_type(16)));
typedef bf16  v8bf  __attribute__((ext_vector_type(8)));
typedef float v8f   __attribute__((ext_vector_type(8)));

#define WMMA_BF16(a, b, c) \
  __builtin_amdgcn_wmma_f32_16x16x32_bf16(false, (a), false, (b), (short)0, (c), false, false)

// ---------------------------------------------------------------------------
// Fragment loaders (wave32; contiguous k order used here, same for A and B)
// A 16x32 bf16: lanes 0-15 -> M=lane, 16 contiguous K starting at 0
//               lanes 16-31 -> M=lane-16, 16 contiguous K starting at 16
// B 32x16 bf16: lanes 0-15 -> N=lane, 16 contiguous K starting at 0
//               lanes 16-31 -> N=lane-16, 16 contiguous K starting at 16
// ---------------------------------------------------------------------------
__device__ __forceinline__ v16bf load_frag_a(const bf16* __restrict__ p) {
  v8bf lo = *(const v8bf*)(p);
  v8bf hi = *(const v8bf*)(p + 8);
  v16bf r;
#pragma unroll
  for (int i = 0; i < 8; ++i) { r[i] = lo[i]; r[i + 8] = hi[i]; }
  return r;
}

__device__ __forceinline__ v16bf load_frag_b(const bf16* __restrict__ p) {
  v8bf lo = *(const v8bf*)(p);
  v8bf hi = *(const v8bf*)(p + 8);
  v16bf r;
#pragma unroll
  for (int i = 0; i < 8; ++i) { r[i] = lo[i]; r[i + 8] = hi[i]; }
  return r;
}

// ---------------------------------------------------------------------------
// f32 -> bf16 convert
// ---------------------------------------------------------------------------
__global__ void k_f32_to_bf16(const float* __restrict__ in, bf16* __restrict__ out, int n) {
  int i = blockIdx.x * blockDim.x + threadIdx.x;
  if (i < n) out[i] = (bf16)in[i];
}

// ---------------------------------------------------------------------------
// Fused GEMM: u = x@Wi^T + bi ; g = sigmoid(x@Wg^T + bg)
// writes g (bf16) and ub = (1-g)*u (bf16)
// Block = 256 threads = 8 waves.
// Wave tile = 64(M) x 32(N): 4 M-tiles x 2 N-tiles x 2 weights = 16 WMMA accums.
// 8 waves arranged 2(M) x 4(N) -> block tile 128(M) x 128(N).
// grid = (N/128, M/128). 16 WMMAs per 16 b128 loads per K-step.
// ---------------------------------------------------------------------------
__global__ __launch_bounds__(256) void k_gemm_ug(
    const bf16* __restrict__ xb,
    const bf16* __restrict__ wib, const bf16* __restrict__ wgb,
    const float* __restrict__ bi, const float* __restrict__ bg,
    bf16* __restrict__ g_out, bf16* __restrict__ ub_out)
{
  constexpr int K = D_;
  constexpr int N = D_;
  const int lane  = threadIdx.x & 31;
  const int wave  = threadIdx.x >> 5;
  const int waveM = wave & 1;        // 0..1 -> 64 rows each
  const int waveN = wave >> 1;       // 0..3 -> 32 cols each
  const int m0 = blockIdx.y * 128 + waveM * 64;
  const int n0 = blockIdx.x * 128 + waveN * 32;
  const int lmod  = lane & 15;
  const int lhalf = lane >> 4;
  const int aoff  = lhalf ? 16 : 0;  // K offset for A upper lane half
  const int boff  = lhalf ? 16 : 0;  // K offset for B upper lane half

  v8f accU[4][2], accG[4][2];
#pragma unroll
  for (int mi = 0; mi < 4; ++mi)
#pragma unroll
    for (int ni = 0; ni < 2; ++ni) { accU[mi][ni] = (v8f)0.0f; accG[mi][ni] = (v8f)0.0f; }

  const bf16* arow[4];
#pragma unroll
  for (int mi = 0; mi < 4; ++mi)
    arow[mi] = xb + (size_t)(m0 + 16 * mi + lmod) * K + aoff;
  const bf16* bwi[2];
  const bf16* bwg[2];
#pragma unroll
  for (int ni = 0; ni < 2; ++ni) {
    bwi[ni] = wib + (size_t)(n0 + 16 * ni + lmod) * K + boff;
    bwg[ni] = wgb + (size_t)(n0 + 16 * ni + lmod) * K + boff;
  }

  for (int k0 = 0; k0 < K; k0 += 32) {
    v16bf a[4];
#pragma unroll
    for (int mi = 0; mi < 4; ++mi) a[mi] = load_frag_a(arow[mi] + k0);
    v16bf wiF[2], wgF[2];
#pragma unroll
    for (int ni = 0; ni < 2; ++ni) {
      wiF[ni] = load_frag_b(bwi[ni] + k0);
      wgF[ni] = load_frag_b(bwg[ni] + k0);
    }
#pragma unroll
    for (int mi = 0; mi < 4; ++mi) {
#pragma unroll
      for (int ni = 0; ni < 2; ++ni) {
        accU[mi][ni] = WMMA_BF16(a[mi], wiF[ni], accU[mi][ni]);
        accG[mi][ni] = WMMA_BF16(a[mi], wgF[ni], accG[mi][ni]);
      }
    }
  }

  // Epilogue: bias, sigmoid, (1-g)*u, store bf16
#pragma unroll
  for (int ni = 0; ni < 2; ++ni) {
    const int col = n0 + 16 * ni + lmod;
    const float biN = bi[col];
    const float bgN = bg[col];
#pragma unroll
    for (int mi = 0; mi < 4; ++mi) {
      v8f cu = accU[mi][ni];
      v8f cg = accG[mi][ni];
#pragma unroll
      for (int v = 0; v < 8; ++v) {
        const int row = m0 + 16 * mi + 8 * lhalf + v;
        const float uu = cu[v] + biN;
        const float gl = cg[v] + bgN;
        const float gv = 1.0f / (1.0f + __expf(-gl));
        const float ub = (1.0f - gv) * uu;
        const size_t o = (size_t)row * N + col;
        g_out[o]  = (bf16)gv;
        ub_out[o] = (bf16)ub;
      }
    }
  }
}

// ---------------------------------------------------------------------------
// Chunked scan, pass 1: per (b, chunk, d) compute P = prod g, S = local scan
// threads: B*NC*D ; i = (b*NC + c)*D + d
// ---------------------------------------------------------------------------
__global__ __launch_bounds__(256) void k_scan_p1(
    const bf16* __restrict__ g, const bf16* __restrict__ ub,
    float* __restrict__ P, float* __restrict__ S)
{
  const int i = blockIdx.x * blockDim.x + threadIdx.x;
  const int d = i % D_;
  const int tmp = i / D_;
  const int c = tmp % NC_;
  const int b = tmp / NC_;
  size_t base = ((size_t)b * T_ + (size_t)c * CL_) * D_ + d;
  float p = 1.0f, s = 0.0f;
#pragma unroll 8
  for (int t = 0; t < CL_; ++t) {
    if ((t & 7) == 0) {  // prefetch ~32 timesteps (64KB) ahead
      __builtin_prefetch(g + base + 32 * (size_t)D_, 0, 0);
      __builtin_prefetch(ub + base + 32 * (size_t)D_, 0, 0);
    }
    const float gv = (float)g[base];
    const float uv = (float)ub[base];
    s = gv * s + uv;
    p *= gv;
    base += D_;
  }
  P[i] = p;
  S[i] = s;
}

// ---------------------------------------------------------------------------
// Pass 2: sequential combine across NC chunks -> carry-in per chunk
// threads: B*D ; j = b*D + d
// ---------------------------------------------------------------------------
__global__ __launch_bounds__(256) void k_scan_p2(
    const float* __restrict__ P, const float* __restrict__ S, float* __restrict__ Hin)
{
  const int j = blockIdx.x * blockDim.x + threadIdx.x;
  const int d = j % D_;
  const int b = j / D_;
  float hin = 0.0f;
#pragma unroll
  for (int c = 0; c < NC_; ++c) {
    const int idx = (b * NC_ + c) * D_ + d;
    Hin[idx] = hin;
    hin = P[idx] * hin + S[idx];
  }
}

// ---------------------------------------------------------------------------
// Pass 3: rescan each chunk with its carry, emit h as bf16 (GEMM-ready)
// ---------------------------------------------------------------------------
__global__ __launch_bounds__(256) void k_scan_p3(
    const bf16* __restrict__ g, const bf16* __restrict__ ub,
    const float* __restrict__ Hin, bf16* __restrict__ hb)
{
  const int i = blockIdx.x * blockDim.x + threadIdx.x;
  const int d = i % D_;
  const int tmp = i / D_;
  const int c = tmp % NC_;
  const int b = tmp / NC_;
  size_t base = ((size_t)b * T_ + (size_t)c * CL_) * D_ + d;
  float h = Hin[i];
#pragma unroll 8
  for (int t = 0; t < CL_; ++t) {
    if ((t & 7) == 0) {
      __builtin_prefetch(g + base + 32 * (size_t)D_, 0, 0);
      __builtin_prefetch(ub + base + 32 * (size_t)D_, 0, 0);
    }
    const float gv = (float)g[base];
    const float uv = (float)ub[base];
    h = gv * h + uv;
    hb[base] = (bf16)h;
    base += D_;
  }
}

// ---------------------------------------------------------------------------
// Output GEMM: out = h @ Wo^T + bo   (f32 output)
// Wave tile = 64(M) x 32(N): 4 M-tiles x 2 N-tiles = 8 WMMA accums.
// 8 WMMAs per 12 b128 loads per K-step; block tile 128x128, grid (N/128, M/128).
// ---------------------------------------------------------------------------
__global__ __launch_bounds__(256) void k_gemm_o(
    const bf16* __restrict__ hb, const bf16* __restrict__ wob,
    const float* __restrict__ bo, float* __restrict__ out)
{
  constexpr int K = D_;
  constexpr int N = D_;
  const int lane  = threadIdx.x & 31;
  const int wave  = threadIdx.x >> 5;
  const int waveM = wave & 1;
  const int waveN = wave >> 1;
  const int m0 = blockIdx.y * 128 + waveM * 64;
  const int n0 = blockIdx.x * 128 + waveN * 32;
  const int lmod  = lane & 15;
  const int lhalf = lane >> 4;
  const int aoff  = lhalf ? 16 : 0;
  const int boff  = lhalf ? 16 : 0;

  v8f acc[4][2];
#pragma unroll
  for (int mi = 0; mi < 4; ++mi)
#pragma unroll
    for (int ni = 0; ni < 2; ++ni) acc[mi][ni] = (v8f)0.0f;

  const bf16* arow[4];
#pragma unroll
  for (int mi = 0; mi < 4; ++mi)
    arow[mi] = hb + (size_t)(m0 + 16 * mi + lmod) * K + aoff;
  const bf16* bw[2];
#pragma unroll
  for (int ni = 0; ni < 2; ++ni)
    bw[ni] = wob + (size_t)(n0 + 16 * ni + lmod) * K + boff;

  for (int k0 = 0; k0 < K; k0 += 32) {
    v16bf a[4];
#pragma unroll
    for (int mi = 0; mi < 4; ++mi) a[mi] = load_frag_a(arow[mi] + k0);
    v16bf bf[2];
#pragma unroll
    for (int ni = 0; ni < 2; ++ni) bf[ni] = load_frag_b(bw[ni] + k0);
#pragma unroll
    for (int mi = 0; mi < 4; ++mi)
#pragma unroll
      for (int ni = 0; ni < 2; ++ni)
        acc[mi][ni] = WMMA_BF16(a[mi], bf[ni], acc[mi][ni]);
  }

#pragma unroll
  for (int ni = 0; ni < 2; ++ni) {
    const int col = n0 + 16 * ni + lmod;
    const float boN = bo[col];
#pragma unroll
    for (int mi = 0; mi < 4; ++mi) {
      v8f c = acc[mi][ni];
#pragma unroll
      for (int v = 0; v < 8; ++v) {
        const int row = m0 + 16 * mi + 8 * lhalf + v;
        out[(size_t)row * N + col] = c[v] + boN;
      }
    }
  }
}

// ---------------------------------------------------------------------------
// Host-side launch
// ---------------------------------------------------------------------------
extern "C" void kernel_launch(void* const* d_in, const int* in_sizes, int n_in,
                              void* d_out, int out_size, void* d_ws, size_t ws_size,
                              hipStream_t stream) {
  (void)in_sizes; (void)n_in; (void)out_size; (void)ws_size;

  const float* x  = (const float*)d_in[0];
  const float* Wi = (const float*)d_in[1];
  const float* bi = (const float*)d_in[2];
  const float* Wg = (const float*)d_in[3];
  const float* bg = (const float*)d_in[4];
  const float* Wo = (const float*)d_in[5];
  const float* bo = (const float*)d_in[6];
  float* out = (float*)d_out;

  // Workspace carve-up (256B aligned)
  char* ws = (char*)d_ws;
  size_t off = 0;
  auto carve = [&](size_t bytes) {
    char* p = ws + off;
    off += (bytes + 255) & ~(size_t)255;
    return p;
  };
  const size_t xElems = (size_t)M_ * D_;      // 33.5M
  const size_t wElems = (size_t)D_ * D_;      // 1M

  bf16*  xb  = (bf16*)carve(xElems * sizeof(bf16));   // 64 MB
  bf16*  wib = (bf16*)carve(wElems * sizeof(bf16));   //  2 MB
  bf16*  wgb = (bf16*)carve(wElems * sizeof(bf16));
  bf16*  wob = (bf16*)carve(wElems * sizeof(bf16));
  bf16*  gb  = (bf16*)carve(xElems * sizeof(bf16));   // 64 MB
  bf16*  ubb = (bf16*)carve(xElems * sizeof(bf16));   // 64 MB
  bf16*  hb  = (bf16*)carve(xElems * sizeof(bf16));   // 64 MB
  float* P   = (float*)carve((size_t)B_ * NC_ * D_ * sizeof(float));
  float* S   = (float*)carve((size_t)B_ * NC_ * D_ * sizeof(float));
  float* Hin = (float*)carve((size_t)B_ * NC_ * D_ * sizeof(float));

  // 1) Convert inputs to bf16
  {
    const int thr = 256;
    k_f32_to_bf16<<<(int)((xElems + thr - 1) / thr), thr, 0, stream>>>(x, xb, (int)xElems);
    k_f32_to_bf16<<<(int)((wElems + thr - 1) / thr), thr, 0, stream>>>(Wi, wib, (int)wElems);
    k_f32_to_bf16<<<(int)((wElems + thr - 1) / thr), thr, 0, stream>>>(Wg, wgb, (int)wElems);
    k_f32_to_bf16<<<(int)((wElems + thr - 1) / thr), thr, 0, stream>>>(Wo, wob, (int)wElems);
  }

  // 2) Fused u/g GEMM + sigmoid + (1-g)*u epilogue
  {
    dim3 grid(D_ / 128, M_ / 128);   // (8, 256)
    k_gemm_ug<<<grid, 256, 0, stream>>>(xb, wib, wgb, bi, bg, gb, ubb);
  }

  // 3) Chunked gated scan
  {
    const int n1 = B_ * NC_ * D_;   // 65536
    const int n2 = B_ * D_;         //  8192
    k_scan_p1<<<n1 / 256, 256, 0, stream>>>(gb, ubb, P, S);
    k_scan_p2<<<n2 / 256, 256, 0, stream>>>(P, S, Hin);
    k_scan_p3<<<n1 / 256, 256, 0, stream>>>(gb, ubb, Hin, hb);
  }

  // 4) Output GEMM
  {
    dim3 grid(D_ / 128, M_ / 128);
    k_gemm_o<<<grid, 256, 0, stream>>>(hb, wob, bo, out);
  }
}
